// BlockModelWrapper_57208964382980
// MI455X (gfx1250) — matrix operand order repaired and checked by reading, not compiled, not run on hardware
//
#include <hip/hip_runtime.h>
#include <cstddef>
#include <cstdint>

// ---------------------------------------------------------------------------
// Problem constants (match reference)
// ---------------------------------------------------------------------------
#define BB      2
#define LL      2048
#define DMODEL  2048
#define DINNER  4096
#define DSTATE  16
#define DTRANK  128
#define DCONV   4
#define TOK     (BB * LL)              // 4096 tokens
#define XPROJ_N (DTRANK + 2 * DSTATE)  // 160

typedef __bf16 bf16x16 __attribute__((ext_vector_type(16)));
typedef __bf16 bf16x8  __attribute__((ext_vector_type(8)));
typedef float  f32x8   __attribute__((ext_vector_type(8)));

// float -> bf16 (round to nearest even), stored as u16 to keep host code clean
__device__ __forceinline__ unsigned short f2bf(float f) {
    unsigned int u = __float_as_uint(f);
    u += 0x7FFFu + ((u >> 16) & 1u);
    return (unsigned short)(u >> 16);
}

__device__ __forceinline__ float silu(float v) {
    return v / (1.0f + __expf(-v));
}

// ---------------------------------------------------------------------------
// Kernel 1: embedding lookup + residual add + RMSNorm
//   residual_out (fp32, -> d_out 2nd half), hs normalized (bf16 for GEMM)
//   one block per token, 256 threads, 8 elements/thread
// ---------------------------------------------------------------------------
__global__ void __launch_bounds__(256)
embed_norm_kernel(const int* __restrict__ ids,
                  const float* __restrict__ residual_in,
                  const float* __restrict__ embed,
                  const float* __restrict__ norm_w,
                  float* __restrict__ residual_out,
                  unsigned short* __restrict__ hsbf) {
    const int t   = blockIdx.x;
    const int tid = threadIdx.x;
    const int id  = ids[t];
    const float* eb = embed + (size_t)id * DMODEL;
    const float* rb = residual_in + (size_t)t * DMODEL;
    float* ro       = residual_out + (size_t)t * DMODEL;

    float v[8];
    float ss = 0.0f;
#pragma unroll
    for (int i = 0; i < 8; ++i) {
        const int d = tid + i * 256;
        const float e = eb[d] + rb[d];
        v[i] = e;
        ss += e * e;
        ro[d] = e;
    }
    __shared__ float red[256];
    red[tid] = ss;
    __syncthreads();
#pragma unroll
    for (int s = 128; s > 0; s >>= 1) {
        if (tid < s) red[tid] += red[tid + s];
        __syncthreads();
    }
    const float inv = rsqrtf(red[0] / (float)DMODEL + 1e-5f);
    unsigned short* ho = hsbf + (size_t)t * DMODEL;
#pragma unroll
    for (int i = 0; i < 8; ++i) {
        const int d = tid + i * 256;
        ho[d] = f2bf(v[i] * inv * norm_w[d]);
    }
}

// ---------------------------------------------------------------------------
// Kernel 2: generic fp32 -> bf16 conversion
// ---------------------------------------------------------------------------
__global__ void __launch_bounds__(256)
cvt_bf16_kernel(const float* __restrict__ src, unsigned short* __restrict__ dst, int n) {
    const int i = blockIdx.x * 256 + threadIdx.x;
    if (i < n) dst[i] = f2bf(src[i]);
}

// ---------------------------------------------------------------------------
// Kernel 3: WMMA bf16 GEMM:  C[M,N] = A[M,K] * W[N,K]^T  (fp32 accumulate)
//   A, W row-major bf16 (stride K).  K % 32 == 0, M % 128 == 0; N may be
//   ragged (handled by clamping the B row index -- branch-free inner loop;
//   out-of-range lanes compute a duplicated valid column that is not stored).
//   Wave tile: 64x32 (4x2 of 16x16x32 WMMA frags -> 8 v_wmma per K step,
//   A reuse 2x, B reuse 4x).  8 waves arranged 2x4 -> 128x128 block tile.
//   A/B fragments: contiguous K halves (k = 16*half + s), same order in both; C/D follows the ISA layout (wave32).
// ---------------------------------------------------------------------------
__global__ void __launch_bounds__(256)
gemm_bf16_wmma(const unsigned short* __restrict__ Abf,
               const unsigned short* __restrict__ Wbf,
               float* __restrict__ C,
               int M, int N, int K) {
    const int lane = threadIdx.x & 31;
    const int wave = threadIdx.x >> 5;   // 0..7
    const int wm   = wave >> 2;          // 0..1
    const int wn   = wave & 3;           // 0..3
    const int m0   = blockIdx.x * 128 + wm * 64;
    const int n0   = blockIdx.y * 128 + wn * 32;
    const int half = lane >> 4;          // 0/1
    const int l16  = lane & 15;
    const int aoff = half << 4;          // A: K half 0 or 16
    const int boff = half << 4;          // B: K half 0 or 16

    // Branch-free ragged-N handling: clamp row index; guard only the store.
    const int nidx[2] = { n0 + l16, n0 + 16 + l16 };
    const unsigned short* brow[2];
#pragma unroll
    for (int j = 0; j < 2; ++j) {
        const int nc = (nidx[j] < N) ? nidx[j] : (N - 1);
        brow[j] = Wbf + (size_t)nc * K + boff;
    }
    const unsigned short* arow[4];
#pragma unroll
    for (int i = 0; i < 4; ++i)
        arow[i] = Abf + (size_t)(m0 + i * 16 + l16) * K + aoff;

    f32x8 acc[4][2] = {};

    for (int k0 = 0; k0 < K; k0 += 32) {
        bf16x16 afrag[4], bfrag[2];
#pragma unroll
        for (int i = 0; i < 4; ++i) {
            // A fragment: 16 contiguous K values (K-half by lane)
            const unsigned short* ap = arow[i] + k0;
            afrag[i] = *(const bf16x16*)ap;
            __builtin_prefetch(ap + 64, 0, 1);
        }
#pragma unroll
        for (int j = 0; j < 2; ++j) {
            // B fragment: 16 contiguous K values from W row n (K-half by lane)
            const unsigned short* bp = brow[j] + k0;
            bfrag[j] = *(const bf16x16*)bp;
            __builtin_prefetch(bp + 64, 0, 1);
        }
#pragma unroll
        for (int i = 0; i < 4; ++i)
#pragma unroll
            for (int j = 0; j < 2; ++j)
                acc[i][j] = __builtin_amdgcn_wmma_f32_16x16x32_bf16(
                    false, afrag[i], false, bfrag[j],
                    (short)0, acc[i][j], false, false);
    }

    // C/D layout: row = base + v + 8*half, col = l16
#pragma unroll
    for (int i = 0; i < 4; ++i) {
        const int mrow = m0 + i * 16 + (half << 3);
#pragma unroll
        for (int j = 0; j < 2; ++j) {
            if (nidx[j] < N) {
#pragma unroll
                for (int v = 0; v < 8; ++v)
                    C[(size_t)(mrow + v) * N + nidx[j]] = acc[i][j][v];
            }
        }
    }
}

// ---------------------------------------------------------------------------
// Kernel 4: causal depthwise conv1d (D_CONV=4) + bias + SiLU
//   reads xc = xz[..., 0:DINNER]; writes x fp32 and x bf16
// ---------------------------------------------------------------------------
__global__ void __launch_bounds__(256)
conv_silu_kernel(const float* __restrict__ xz,
                 const float* __restrict__ conv_w,
                 const float* __restrict__ conv_b,
                 float* __restrict__ x,
                 unsigned short* __restrict__ xbf) {
    const int idx = blockIdx.x * 256 + threadIdx.x;
    if (idx >= TOK * DINNER) return;
    const int d = idx & (DINNER - 1);
    const int t = idx >> 12;            // token index (b*LL + l)
    const int l = t & (LL - 1);
    const int b = t >> 11;

    float acc = conv_b[d];
#pragma unroll
    for (int j = 0; j < DCONV; ++j) {
        const int ls = l - (DCONV - 1) + j;
        if (ls >= 0)
            acc += xz[((size_t)(b * LL + ls)) * (2 * DINNER) + d] * conv_w[d * DCONV + j];
    }
    const float s = silu(acc);
    x[idx] = s;
    xbf[idx] = f2bf(s);
}

// ---------------------------------------------------------------------------
// Kernel 5: extract dt-rank slice of x_dbl (cols 0..127 of 160) -> bf16
// ---------------------------------------------------------------------------
__global__ void __launch_bounds__(256)
dtrank_bf16_kernel(const float* __restrict__ xdbl, unsigned short* __restrict__ dtbf) {
    const int idx = blockIdx.x * 256 + threadIdx.x;
    if (idx >= TOK * DTRANK) return;
    const int r = idx & (DTRANK - 1);
    const int t = idx >> 7;
    dtbf[idx] = f2bf(xdbl[(size_t)t * XPROJ_N + r]);
}

// ---------------------------------------------------------------------------
// Kernel 6: dt = softplus(dt + dt_proj_b)   (in place on fp32 GEMM output)
// ---------------------------------------------------------------------------
__global__ void __launch_bounds__(256)
softplus_bias_kernel(float* __restrict__ dt, const float* __restrict__ bias) {
    const int idx = blockIdx.x * 256 + threadIdx.x;
    if (idx >= TOK * DINNER) return;
    const int d = idx & (DINNER - 1);
    const float v = dt[idx] + bias[d];
    dt[idx] = (v > 20.0f) ? v : log1pf(__expf(v));
}

// ---------------------------------------------------------------------------
// Kernel 7: selective scan, fused with  +x*D  and  *silu(z), bf16 output.
//   One thread per (b, d) channel; 16-state recurrence in registers.
//   B_t / C_t (16 each) broadcast through LDS per timestep.
//   grid = (DINNER/256, BB), block = 256
// ---------------------------------------------------------------------------
__global__ void __launch_bounds__(256)
scan_kernel(const float* __restrict__ dt,     // (TOK, DINNER)
            const float* __restrict__ xdbl,   // (TOK, 160): [128:144)=B, [144:160)=C
            const float* __restrict__ x,      // (TOK, DINNER)
            const float* __restrict__ xz,     // (TOK, 2*DINNER): z at [DINNER+d]
            const float* __restrict__ A_log,  // (DINNER, DSTATE)
            const float* __restrict__ Dp,     // (DINNER)
            unsigned short* __restrict__ ybf) // (TOK, DINNER)
{
    const int tid = threadIdx.x;
    const int d   = blockIdx.x * 256 + tid;
    const int b   = blockIdx.y;

    float a[DSTATE], h[DSTATE];
#pragma unroll
    for (int n = 0; n < DSTATE; ++n) {
        a[n] = -__expf(A_log[(size_t)d * DSTATE + n]);
        h[n] = 0.0f;
    }
    const float Dd = Dp[d];

    __shared__ float sB[DSTATE];
    __shared__ float sC[DSTATE];

    for (int l = 0; l < LL; ++l) {
        const size_t t = (size_t)b * LL + l;
        if (tid < DSTATE)
            sB[tid] = xdbl[t * XPROJ_N + DTRANK + tid];
        else if (tid < 2 * DSTATE)
            sC[tid - DSTATE] = xdbl[t * XPROJ_N + DTRANK + DSTATE + (tid - DSTATE)];
        __syncthreads();

        const float dtv = dt[t * DINNER + d];
        const float xv  = x[t * DINNER + d];
        float acc = 0.0f;
#pragma unroll
        for (int n = 0; n < DSTATE; ++n) {
            const float dA = __expf(dtv * a[n]);
            h[n] = dA * h[n] + dtv * sB[n] * xv;
            acc += h[n] * sC[n];
        }
        const float zv = xz[t * (2 * DINNER) + DINNER + d];
        const float yv = (acc + xv * Dd) * silu(zv);
        ybf[t * DINNER + d] = f2bf(yv);
        __syncthreads();
    }
}

// ---------------------------------------------------------------------------
// Host launcher
// ---------------------------------------------------------------------------
extern "C" void kernel_launch(void* const* d_in, const int* in_sizes, int n_in,
                              void* d_out, int out_size, void* d_ws, size_t ws_size,
                              hipStream_t stream) {
    (void)in_sizes; (void)n_in; (void)out_size; (void)ws_size;

    const int*   ids        = (const int*)  d_in[0];
    // d_in[1] (hidden_states) is overwritten by the embedding lookup -> unused
    const float* residual   = (const float*)d_in[2];
    const float* embed      = (const float*)d_in[3];
    const float* norm_w     = (const float*)d_in[4];
    const float* in_proj_w  = (const float*)d_in[5];   // (8192, 2048)
    const float* conv_w     = (const float*)d_in[6];   // (4096, 4)
    const float* conv_b     = (const float*)d_in[7];
    const float* x_proj_w   = (const float*)d_in[8];   // (160, 4096)
    const float* dt_proj_w  = (const float*)d_in[9];   // (4096, 128)
    const float* dt_proj_b  = (const float*)d_in[10];
    const float* A_log      = (const float*)d_in[11];  // (4096, 16)
    const float* D_param    = (const float*)d_in[12];
    const float* out_proj_w = (const float*)d_in[13];  // (2048, 4096)

    float* out_hs  = (float*)d_out;                              // (TOK, DMODEL)
    float* out_res = (float*)d_out + (size_t)TOK * DMODEL;       // (TOK, DMODEL)

    // ---- workspace carve-up (bytes, all 256-aligned) ----
    unsigned char* ws = (unsigned char*)d_ws;
    size_t off = 0;
    auto carve = [&](size_t bytes) { void* p = ws + off; off += (bytes + 255) & ~(size_t)255; return p; };

    unsigned short* w1bf  = (unsigned short*)carve((size_t)2 * DINNER * DMODEL * 2); // in_proj bf16
    unsigned short* wobf  = (unsigned short*)carve((size_t)DMODEL * DINNER * 2);     // out_proj bf16
    unsigned short* wxbf  = (unsigned short*)carve((size_t)XPROJ_N * DINNER * 2);    // x_proj bf16
    unsigned short* wdtbf = (unsigned short*)carve((size_t)DINNER * DTRANK * 2);     // dt_proj bf16
    unsigned short* hsbf  = (unsigned short*)carve((size_t)TOK * DMODEL * 2);        // normed hs bf16
    float*          xz    = (float*)         carve((size_t)TOK * 2 * DINNER * 4);    // in_proj out
    float*          xbuf  = (float*)         carve((size_t)TOK * DINNER * 4);        // conv+silu fp32
    unsigned short* xbf   = (unsigned short*)carve((size_t)TOK * DINNER * 2);        // conv+silu bf16
    float*          xdbl  = (float*)         carve((size_t)TOK * XPROJ_N * 4);       // x_proj out
    unsigned short* dtbf  = (unsigned short*)carve((size_t)TOK * DTRANK * 2);        // dt-rank bf16
    float*          dtbuf = (float*)         carve((size_t)TOK * DINNER * 4);        // dt fp32
    unsigned short* ybf   = (unsigned short*)carve((size_t)TOK * DINNER * 2);        // gated y bf16

    // 1) embed + residual + rmsnorm (also writes residual output)
    embed_norm_kernel<<<TOK, 256, 0, stream>>>(ids, residual, embed, norm_w, out_res, hsbf);

    // 2) weight conversions fp32 -> bf16
    {
        int n;
        n = 2 * DINNER * DMODEL;
        cvt_bf16_kernel<<<(n + 255) / 256, 256, 0, stream>>>(in_proj_w, w1bf, n);
        n = DMODEL * DINNER;
        cvt_bf16_kernel<<<(n + 255) / 256, 256, 0, stream>>>(out_proj_w, wobf, n);
        n = XPROJ_N * DINNER;
        cvt_bf16_kernel<<<(n + 255) / 256, 256, 0, stream>>>(x_proj_w, wxbf, n);
        n = DINNER * DTRANK;
        cvt_bf16_kernel<<<(n + 255) / 256, 256, 0, stream>>>(dt_proj_w, wdtbf, n);
    }

    // 3) xz = hs @ in_proj_w^T      (M=4096, N=8192, K=2048)
    gemm_bf16_wmma<<<dim3(TOK / 128, (2 * DINNER + 127) / 128), 256, 0, stream>>>(
        hsbf, w1bf, xz, TOK, 2 * DINNER, DMODEL);

    // 4) x = silu(conv1d(xc) + b)
    {
        const int n = TOK * DINNER;
        conv_silu_kernel<<<(n + 255) / 256, 256, 0, stream>>>(xz, conv_w, conv_b, xbuf, xbf);
    }

    // 5) x_dbl = x @ x_proj_w^T     (M=4096, N=160, K=4096)
    gemm_bf16_wmma<<<dim3(TOK / 128, (XPROJ_N + 127) / 128), 256, 0, stream>>>(
        xbf, wxbf, xdbl, TOK, XPROJ_N, DINNER);

    // 6) dt-rank slice -> bf16
    {
        const int n = TOK * DTRANK;
        dtrank_bf16_kernel<<<(n + 255) / 256, 256, 0, stream>>>(xdbl, dtbf);
    }

    // 7) dt = dtrank @ dt_proj_w^T  (M=4096, N=4096, K=128)
    gemm_bf16_wmma<<<dim3(TOK / 128, (DINNER + 127) / 128), 256, 0, stream>>>(
        dtbf, wdtbf, dtbuf, TOK, DINNER, DTRANK);

    // 8) dt = softplus(dt + bias)
    {
        const int n = TOK * DINNER;
        softplus_bias_kernel<<<(n + 255) / 256, 256, 0, stream>>>(dtbuf, dt_proj_b);
    }

    // 9) selective scan (+ x*D, * silu(z)) -> y bf16
    scan_kernel<<<dim3(DINNER / 256, BB), 256, 0, stream>>>(
        dtbuf, xdbl, xbuf, xz, A_log, D_param, ybf);

    // 10) out = y @ out_proj_w^T    (M=4096, N=2048, K=4096) -> d_out first half
    gemm_bf16_wmma<<<dim3(TOK / 128, (DMODEL + 127) / 128), 256, 0, stream>>>(
        ybf, wobf, out_hs, TOK, DMODEL, DINNER);
}
